// GCN_87325275062334
// MI455X (gfx1250) — matrix operand order repaired and checked by reading, not compiled, not run on hardware
//
#include <hip/hip_runtime.h>

// ---------------------------------------------------------------------------
// L=2 layers, H=8 heads, D=128, DFF=512, B=16, M=1024, N=16384 tokens/nodes,
// E=524288 edges. All dense math via v_wmma_f32_16x16x32_bf16 (bf16 operands,
// f32 accumulate). Activations are pre-converted to bf16 once per GEMM so the
// WMMA loop issues no converts and streams half the bytes. Each GEMM wave
// register-blocks 4 M-tiles (64x16 output) to reuse the B operand 4x.
// ---------------------------------------------------------------------------

#define DM    128
#define DFFN  512
#define HEADS 8
#define DH    16
#define NTOK  16384
#define MSEQ  1024
#define LAYERS 2

typedef __attribute__((ext_vector_type(16))) __bf16 v16bf;
typedef __attribute__((ext_vector_type(8)))  __bf16 v8bf;
typedef __attribute__((ext_vector_type(8)))  float  v8f;

enum GemmMode { OUT_F32 = 0, OUT_F32_RELU = 1, OUT_F32_RESID = 2,
                OUT_BF16_QK = 3, OUT_BF16_VT = 4 };

// ---------------------------------------------------------------------------
// fp32 -> bf16 streaming converts
// ---------------------------------------------------------------------------
__global__ void act_convert(const float* __restrict__ x, __bf16* __restrict__ y,
                            int total) {
  const int idx = (blockIdx.x * blockDim.x + threadIdx.x) * 4;
  if (idx >= total) return;
  const float4 v = *(const float4*)(x + idx);
  y[idx + 0] = (__bf16)v.x;
  y[idx + 1] = (__bf16)v.y;
  y[idx + 2] = (__bf16)v.z;
  y[idx + 3] = (__bf16)v.w;
}

// W[K][N] fp32 -> Wt[N][K] bf16 (B-operand becomes one 32B load per lane)
__global__ void wt_convert(const float* __restrict__ W, __bf16* __restrict__ Wt,
                           int K, int N) {
  const int idx = blockIdx.x * blockDim.x + threadIdx.x;
  if (idx >= K * N) return;
  const int k = idx / N, n = idx % N;
  Wt[(size_t)n * K + k] = (__bf16)W[idx];
}

// ---------------------------------------------------------------------------
// GEMM: out[M][N] = A[M][K] @ W[K][N] (+bias)(+resid)(relu).
// One wave per 64x16 output block: 4 M-tiles register-blocked, B reused 4x.
// A-operand (16-bit A 16x32, wave32):
//   lanes 0-15 : elems 0-7 -> K=0..7,  elems 8-15 -> K=8..15
//   lanes16-31 : elems 0-7 -> K=16..23, elems 8-15 -> K=24..31
// B-operand (32x16): lanes 0-15 K=0..15, lanes 16-31 K=16..31, N=lane&15.
// C layout: VGPR r, lane L -> D[r + (L<16?0:8)][L&15].
// ---------------------------------------------------------------------------
__global__ __launch_bounds__(32) void gemm_wmma_bf16(
    const __bf16* __restrict__ A, const __bf16* __restrict__ Wt,
    const float* __restrict__ bias, const float* __restrict__ resid,
    void* __restrict__ outp, int M, int N, int K, int mode) {
  const int tm = blockIdx.x;        // 64-row block
  const int tn = blockIdx.y;        // 16-col block
  const int lane = threadIdx.x;
  const int half = lane >> 4;
  const int l16  = lane & 15;
  const int aoff = half ? 16 : 0;
  const int boff = half ? 16 : 0;

  const __bf16* Wrow = Wt + (size_t)(tn * 16 + l16) * K;
  const __bf16* Ar0 = A + (size_t)(tm * 64 +  0 + l16) * K;
  const __bf16* Ar1 = A + (size_t)(tm * 64 + 16 + l16) * K;
  const __bf16* Ar2 = A + (size_t)(tm * 64 + 32 + l16) * K;
  const __bf16* Ar3 = A + (size_t)(tm * 64 + 48 + l16) * K;

  v8f c0 = {}, c1 = {}, c2 = {}, c3 = {};
  for (int k0 = 0; k0 < K; k0 += 32) {
    const v16bf b = *(const v16bf*)(Wrow + k0 + boff);  // reused by 4 tiles
#define GEMM_MT(AR, CC)                                                     \
    {                                                                       \
      v8bf t0 = *(const v8bf*)(AR + k0 + aoff);                             \
      v8bf t1 = *(const v8bf*)(AR + k0 + 8 + aoff);                         \
      v16bf a;                                                              \
      a[0]=t0[0];  a[1]=t0[1];  a[2]=t0[2];  a[3]=t0[3];                    \
      a[4]=t0[4];  a[5]=t0[5];  a[6]=t0[6];  a[7]=t0[7];                    \
      a[8]=t1[0];  a[9]=t1[1];  a[10]=t1[2]; a[11]=t1[3];                   \
      a[12]=t1[4]; a[13]=t1[5]; a[14]=t1[6]; a[15]=t1[7];                   \
      CC = __builtin_amdgcn_wmma_f32_16x16x32_bf16(false, a, false, b,      \
                                                   (short)0, CC, false, false); \
    }
    GEMM_MT(Ar0, c0)
    GEMM_MT(Ar1, c1)
    GEMM_MT(Ar2, c2)
    GEMM_MT(Ar3, c3)
#undef GEMM_MT
  }

  const float bval = bias ? bias[tn * 16 + l16] : 0.0f;
  const int gn = tn * 16 + l16;
#pragma unroll
  for (int mt = 0; mt < 4; ++mt) {
    const v8f c = (mt == 0) ? c0 : (mt == 1) ? c1 : (mt == 2) ? c2 : c3;
#pragma unroll
    for (int r = 0; r < 8; ++r) {
      const int gm = tm * 64 + mt * 16 + r + (half ? 8 : 0);
      float v = c[r] + bval;
      if (mode == OUT_F32_RELU)  v = fmaxf(v, 0.0f);
      if (mode == OUT_F32_RESID) v += resid[(size_t)gm * N + gn];
      if (mode <= OUT_F32_RESID) {
        ((float*)outp)[(size_t)gm * N + gn] = v;
      } else {
        const int bb = gm >> 10, mm = gm & 1023;   // batch, seq pos
        const int hh = gn >> 4,  dd = gn & 15;     // head, head-dim
        __bf16* ob = (__bf16*)outp;
        if (mode == OUT_BF16_QK)      // [b][h][m][d]
          ob[(((size_t)bb * HEADS + hh) * MSEQ + mm) * DH + dd] = (__bf16)v;
        else                          // [b][h][d][m]  (V transposed)
          ob[(((size_t)bb * HEADS + hh) * DH + dd) * MSEQ + mm] = (__bf16)v;
      }
    }
  }
}

// ---------------------------------------------------------------------------
// Flash attention: one wave per (b*H+h, q-tile of 16). dh=16 zero-padded to
// the K=32 bf16 WMMA. S = Q·K^T (WMMA), online softmax with 16-lane shuffle
// reductions, P bounced through 512B LDS into A-layout, O += P·V (WMMA).
// ---------------------------------------------------------------------------
__global__ __launch_bounds__(32) void attn_flash(
    const __bf16* __restrict__ qb, const __bf16* __restrict__ kb,
    const __bf16* __restrict__ vt, float* __restrict__ outp) {
  __shared__ __align__(32) __bf16 pls[16 * 16];
  const int bh   = blockIdx.x;      // b*H + h
  const int qt   = blockIdx.y;      // q tile index
  const int lane = threadIdx.x;
  const int half = lane >> 4;
  const int l16  = lane & 15;
  const float scale = 0.25f;        // 1/sqrt(16)

  // Q as A-operand (d = 0..15 real, 16..31 zero pad)
  v16bf aq = {};
  {
    if (lane < 16)
      aq = *(const v16bf*)(qb + ((size_t)bh * MSEQ + qt * 16 + l16) * DH);
  }

  float mrow[8], lrow[8];
#pragma unroll
  for (int r = 0; r < 8; ++r) { mrow[r] = -1e30f; lrow[r] = 0.0f; }
  v8f oacc = {};

  for (int kt = 0; kt < MSEQ / 16; ++kt) {
    // K tile as B-operand: lane n(=key) holds d=0..15; lanes>=16 are the pad
    v16bf bkv = {};
    if (lane < 16)
      bkv = *(const v16bf*)(kb + ((size_t)bh * MSEQ + kt * 16 + l16) * DH);
    v8f zc = {};
    v8f s = __builtin_amdgcn_wmma_f32_16x16x32_bf16(false, aq, false, bkv,
                                                    (short)0, zc, false, false);
#pragma unroll
    for (int r = 0; r < 8; ++r) {
      float sv = s[r] * scale;
      float mx = sv;                              // row max over 16-lane half
      mx = fmaxf(mx, __shfl_xor(mx, 1));
      mx = fmaxf(mx, __shfl_xor(mx, 2));
      mx = fmaxf(mx, __shfl_xor(mx, 4));
      mx = fmaxf(mx, __shfl_xor(mx, 8));
      float nm    = fmaxf(mrow[r], mx);
      float alpha = __expf(mrow[r] - nm);
      float p     = __expf(sv - nm);
      float ps = p;                               // row sum
      ps += __shfl_xor(ps, 1);
      ps += __shfl_xor(ps, 2);
      ps += __shfl_xor(ps, 4);
      ps += __shfl_xor(ps, 8);
      lrow[r] = lrow[r] * alpha + ps;
      oacc[r] = oacc[r] * alpha;
      mrow[r] = nm;
      pls[(r + half * 8) * 16 + l16] = (__bf16)p; // C-layout -> LDS [m][k]
    }
    // P as A-operand (single wave: DS ops are in program order, no barrier)
    v16bf ap = {};
    {
      if (lane < 16)
        ap = *(const v16bf*)(pls + l16 * 16);
    }
    // V tile as B-operand from transposed layout: lane n(=d) holds k=0..15
    v16bf bvv = {};
    if (lane < 16)
      bvv = *(const v16bf*)(vt + ((size_t)bh * DH + l16) * MSEQ + kt * 16);
    oacc = __builtin_amdgcn_wmma_f32_16x16x32_bf16(false, ap, false, bvv,
                                                   (short)0, oacc, false, false);
  }

  const int bb = bh >> 3, hh = bh & 7;
#pragma unroll
  for (int r = 0; r < 8; ++r) {
    const int m = qt * 16 + r + half * 8;
    outp[((size_t)(bb * MSEQ + m)) * DM + hh * DH + l16] = oacc[r] / lrow[r];
  }
}

// ---------------------------------------------------------------------------
// LayerNorm over rows of 128: one wave per row (4 floats/lane, shuffle reduce)
// ---------------------------------------------------------------------------
__global__ __launch_bounds__(256) void layernorm_k(
    const float* __restrict__ x, const float* __restrict__ g,
    const float* __restrict__ b, float* __restrict__ y) {
  const int wid  = threadIdx.x >> 5;
  const int lane = threadIdx.x & 31;
  const int row  = blockIdx.x * 8 + wid;
  const float4 xv = ((const float4*)(x + (size_t)row * DM))[lane];
  float s = xv.x + xv.y + xv.z + xv.w;
  s += __shfl_xor(s, 1);  s += __shfl_xor(s, 2);  s += __shfl_xor(s, 4);
  s += __shfl_xor(s, 8);  s += __shfl_xor(s, 16);
  const float mu = s * (1.0f / DM);
  const float4 dx = make_float4(xv.x - mu, xv.y - mu, xv.z - mu, xv.w - mu);
  float v = dx.x * dx.x + dx.y * dx.y + dx.z * dx.z + dx.w * dx.w;
  v += __shfl_xor(v, 1);  v += __shfl_xor(v, 2);  v += __shfl_xor(v, 4);
  v += __shfl_xor(v, 8);  v += __shfl_xor(v, 16);
  const float rs = rsqrtf(v * (1.0f / DM) + 1e-5f);
  const float4 gv = ((const float4*)g)[lane];
  const float4 bv = ((const float4*)b)[lane];
  const float4 o = make_float4(dx.x * rs * gv.x + bv.x, dx.y * rs * gv.y + bv.y,
                               dx.z * rs * gv.z + bv.z, dx.w * rs * gv.w + bv.w);
  ((float4*)(y + (size_t)row * DM))[lane] = o;
}

// ---------------------------------------------------------------------------
// GCN pieces: degree (with self loop), D^-1/2, self-loop init, edge scatter
// (f32 atomics), bias+relu finalize.
// ---------------------------------------------------------------------------
__global__ void deg_init_k(float* deg) {
  deg[blockIdx.x * blockDim.x + threadIdx.x] = 1.0f;  // self loop
}
__global__ void deg_acc_k(const int* __restrict__ dst, float* deg, int E) {
  const int t = blockIdx.x * blockDim.x + threadIdx.x;
  if (t < E) atomicAdd(&deg[dst[t]], 1.0f);
}
__global__ void dinv_k(float* deg) {
  const int t = blockIdx.x * blockDim.x + threadIdx.x;
  deg[t] = rsqrtf(fmaxf(deg[t], 1e-12f));
}
__global__ void gcn_self_k(const float* __restrict__ h, const float* __restrict__ dinv,
                           float* __restrict__ acc) {
  const int idx = blockIdx.x * blockDim.x + threadIdx.x;
  const float di = dinv[idx >> 7];
  acc[idx] = h[idx] * di * di;
}
__global__ void gcn_scatter_k(const int* __restrict__ src, const int* __restrict__ dst,
                              const float* __restrict__ h, const float* __restrict__ dinv,
                              float* __restrict__ acc, int E) {
  const int t = blockIdx.x * blockDim.x + threadIdx.x;
  const int e = t >> 3;
  if (e >= E) return;
  const int part = t & 7;                 // 8 threads/edge, 16 floats each
  const int si = src[e], di = dst[e];
  const float w = dinv[si] * dinv[di];
  const float4* hp = (const float4*)(h + (size_t)si * DM + part * 16);
  float* ap = acc + (size_t)di * DM + part * 16;
#pragma unroll
  for (int j = 0; j < 4; ++j) {
    const float4 hv = hp[j];
    atomicAdd(ap + 4 * j + 0, hv.x * w);
    atomicAdd(ap + 4 * j + 1, hv.y * w);
    atomicAdd(ap + 4 * j + 2, hv.z * w);
    atomicAdd(ap + 4 * j + 3, hv.w * w);
  }
}
__global__ void gcn_final_k(const float* __restrict__ acc, const float* __restrict__ bias,
                            float* __restrict__ out) {
  const int idx = blockIdx.x * blockDim.x + threadIdx.x;
  out[idx] = fmaxf(acc[idx] + bias[idx & (DM - 1)], 0.0f);
}

// ---------------------------------------------------------------------------
// Host orchestration. Workspace needs ~118 MB.
// ---------------------------------------------------------------------------
extern "C" void kernel_launch(void* const* d_in, const int* in_sizes, int n_in,
                              void* d_out, int out_size, void* d_ws, size_t ws_size,
                              hipStream_t stream) {
  (void)n_in; (void)out_size; (void)ws_size;
  const float* enc_out = (const float*)d_in[0];
  // d_in[1] (x_enc) is unused by the reference.
  const float* x_raw   = (const float*)d_in[2];
  const int*   eidx    = (const int*)d_in[3];
  const float* conv1_W = (const float*)d_in[4];
  const float* conv1_b = (const float*)d_in[5];
  const float* conv2_W = (const float*)d_in[6];
  const float* conv2_b = (const float*)d_in[7];
  const float* Wq = (const float*)d_in[8];   const float* bq = (const float*)d_in[9];
  const float* Wk = (const float*)d_in[10];  const float* bk = (const float*)d_in[11];
  const float* Wv = (const float*)d_in[12];  const float* bv = (const float*)d_in[13];
  const float* Wo = (const float*)d_in[14];  const float* bo = (const float*)d_in[15];
  const float* W1 = (const float*)d_in[16];  const float* b1 = (const float*)d_in[17];
  const float* W2 = (const float*)d_in[18];  const float* b2 = (const float*)d_in[19];
  const float* ln1_g = (const float*)d_in[20]; const float* ln1_b = (const float*)d_in[21];
  const float* ln2_g = (const float*)d_in[22]; const float* ln2_b = (const float*)d_in[23];

  const int E = in_sizes[3] / 2;
  const int* src = eidx;
  const int* dst = eidx + E;

  // Bump-allocate workspace
  char* ws = (char*)d_ws;
  size_t off = 0;
  auto alloc = [&](size_t bytes) -> void* {
    void* p = ws + off;
    off += (bytes + 255) & ~(size_t)255;
    return p;
  };
  float*  h_cur    = (float*)alloc((size_t)NTOK * DM * 4);
  float*  tmp      = (float*)alloc((size_t)NTOK * DM * 4);
  float*  kv_x     = (float*)alloc((size_t)NTOK * DM * 4);
  float*  g_h      = (float*)alloc((size_t)NTOK * DM * 4);
  float*  g_acc    = (float*)alloc((size_t)NTOK * DM * 4);
  float*  x1       = (float*)alloc((size_t)NTOK * DM * 4);
  __bf16* qb       = (__bf16*)alloc((size_t)NTOK * DM * 2);
  __bf16* kbuf     = (__bf16*)alloc((size_t)NTOK * DM * 2);
  __bf16* vt       = (__bf16*)alloc((size_t)NTOK * DM * 2);
  float*  attn_out = (float*)alloc((size_t)NTOK * DM * 4);
  float*  ffn1     = (float*)alloc((size_t)NTOK * DFFN * 4);
  float*  dinv     = (float*)alloc((size_t)NTOK * 4);
  __bf16* abuf     = (__bf16*)alloc((size_t)NTOK * DFFN * 2);   // bf16 activations
  __bf16* wt_buf   = (__bf16*)alloc((size_t)DFFN * DM * 2);     // bf16 weightsT

  auto run_gemm = [&](const float* Ap, const float* Wsrc, const float* biasp,
                      const float* residp, void* outp, int M, int N, int K, int mode) {
    const int atot = M * K;
    act_convert<<<dim3((atot / 4 + 255) / 256), dim3(256), 0, stream>>>(Ap, abuf, atot);
    const int wtot = K * N;
    wt_convert<<<dim3((wtot + 255) / 256), dim3(256), 0, stream>>>(Wsrc, wt_buf, K, N);
    gemm_wmma_bf16<<<dim3(M / 64, N / 16), dim3(32), 0, stream>>>(
        abuf, wt_buf, biasp, residp, outp, M, N, K, mode);
  };

  // ---- GCN degree normalization (shared by both convs) ----
  deg_init_k<<<dim3(NTOK / 256), dim3(256), 0, stream>>>(dinv);
  deg_acc_k<<<dim3((E + 255) / 256), dim3(256), 0, stream>>>(dst, dinv, E);
  dinv_k<<<dim3(NTOK / 256), dim3(256), 0, stream>>>(dinv);

  auto run_gcn = [&](const float* xin, const float* Wsrc, const float* bsrc, float* xout) {
    run_gemm(xin, Wsrc, nullptr, nullptr, g_h, NTOK, DM, DM, OUT_F32);
    gcn_self_k<<<dim3(NTOK * DM / 256), dim3(256), 0, stream>>>(g_h, dinv, g_acc);
    gcn_scatter_k<<<dim3(((size_t)E * 8 + 255) / 256), dim3(256), 0, stream>>>(
        src, dst, g_h, dinv, g_acc, E);
    gcn_final_k<<<dim3(NTOK * DM / 256), dim3(256), 0, stream>>>(g_acc, bsrc, xout);
  };

  run_gcn(x_raw, conv1_W, conv1_b, x1);    // conv1 -> relu -> x1
  run_gcn(x1,    conv2_W, conv2_b, kv_x);  // conv2 -> relu -> kv

  // ---- Encoder ----
  hipMemcpyAsync(h_cur, enc_out, (size_t)NTOK * DM * 4, hipMemcpyDeviceToDevice, stream);

  for (int l = 0; l < LAYERS; ++l) {
    const size_t wo = (size_t)l * DM * DM;
    const size_t bo1 = (size_t)l * DM;
    // Q/K/V projections directly into attention-friendly bf16 layouts
    run_gemm(h_cur, Wq + wo, bq + bo1, nullptr, qb,   NTOK, DM, DM, OUT_BF16_QK);
    run_gemm(kv_x,  Wk + wo, bk + bo1, nullptr, kbuf, NTOK, DM, DM, OUT_BF16_QK);
    run_gemm(kv_x,  Wv + wo, bv + bo1, nullptr, vt,   NTOK, DM, DM, OUT_BF16_VT);
    // Flash attention: 128 (b,h) x 64 q-tiles
    attn_flash<<<dim3(HEADS * 16, MSEQ / 16), dim3(32), 0, stream>>>(qb, kbuf, vt, attn_out);
    // Output projection + residual, then LN1
    run_gemm(attn_out, Wo + wo, bo + bo1, h_cur, tmp, NTOK, DM, DM, OUT_F32_RESID);
    layernorm_k<<<dim3(NTOK / 8), dim3(256), 0, stream>>>(tmp, ln1_g + bo1, ln1_b + bo1, h_cur);
    // FFN
    run_gemm(h_cur, W1 + (size_t)l * DM * DFFN, b1 + (size_t)l * DFFN, nullptr,
             ffn1, NTOK, DFFN, DM, OUT_F32_RELU);
    run_gemm(ffn1, W2 + (size_t)l * DFFN * DM, b2 + bo1, h_cur,
             tmp, NTOK, DM, DFFN, OUT_F32_RESID);
    layernorm_k<<<dim3(NTOK / 8), dim3(256), 0, stream>>>(tmp, ln2_g + bo1, ln2_b + bo1, h_cur);
  }

  hipMemcpyAsync(d_out, h_cur, (size_t)NTOK * DM * 4, hipMemcpyDeviceToDevice, stream);
}
